// DensityPropCNN_50783693308289
// MI455X (gfx1250) — compile-verified, matrix operand order checked by reading, not run on hardware
//
#include <hip/hip_runtime.h>
#include <hip/hip_bf16.h>
#include <math.h>

typedef __attribute__((ext_vector_type(16))) _Float16 v16h;
typedef __attribute__((ext_vector_type(8)))  _Float16 v8h;
typedef __attribute__((ext_vector_type(8)))  float    v8f;

#define K_NUM 32
#define FEAT 4608          // 32*12*12
#define UNITS 10
#define IMG 784            // 28*28
#define KCHUNKS (FEAT/32)  // 144

__device__ __forceinline__ float softplusf(float x) { return log1pf(__expf(x)); }
__device__ __forceinline__ float cleanf(float x) {
  x = (x != x) ? 1e-5f : x;            // NaN -> 1e-5
  x = __builtin_isinf(x) ? 1.0f : x;   // Inf -> 1.0
  return x;
}

// ---------------------------------------------------------------------------
// WMMA operand layout helpers (ISA 7.12.2, v_wmma_f32_16x16x32_f16):
//  A (16x32, M x K): lane<16 row M=lane  holds K = {c+0..7,  c+16..23}
//                    lane>=16 row M=l-16 holds K = {c+8..15, c+24..31}
//  B (32x16, K x N): lane<16 col N=lane  holds same K split; mirrored.
//  C/D (16x16):      VGPR v -> row (lane<16 ? v : v+8), col = lane%16
// ---------------------------------------------------------------------------
__device__ __forceinline__ int b_klo(int lane, int e) {
  return (lane < 16) ? ((e < 8) ? e : (8 + e))
                     : ((e < 8) ? (8 + e) : (16 + e));
}

// ---------------------------------------------------------------------------
// Kernel 0: pack fc_w_mu [4608,10] (row-major f32) into WMMA B-operand layout.
// Storage: i = (kc*32 + lane)*16 + e  (f16).  N padded 10 -> 16 with zeros.
// Bsq holds w*w for the sigma path.
// ---------------------------------------------------------------------------
__global__ void pack_fc_kernel(const float* __restrict__ fcw,
                               _Float16* __restrict__ Bmu,
                               _Float16* __restrict__ Bsq) {
  int gid = blockIdx.x * blockDim.x + threadIdx.x;
  const int total = KCHUNKS * 32 * 16;
  for (int i = gid; i < total; i += gridDim.x * blockDim.x) {
    int e    = i & 15;
    int lane = (i >> 4) & 31;
    int kc   = i >> 9;
    int n    = lane & 15;
    int k    = kc * 32 + b_klo(lane, e);
    float w  = (n < UNITS) ? fcw[k * UNITS + n] : 0.0f;
    Bmu[i] = (_Float16)w;
    Bsq[i] = (_Float16)(w * w);
  }
}

// ---------------------------------------------------------------------------
// Kernel 1: fused conv(5x5,valid) + variance + ReLU + 2x2 maxpool per image,
// conv as implicit GEMM on WMMA:
//   per image: patches [576 x 25] @ W^T [25 x 32], K padded to 32
//   = 36 M-tiles x 2 N-tiles x 1 K-step of v_wmma_f32_16x16x32_f16.
// One block per image, 256 threads (8 waves). Two pool-aligned phases
// (conv rows Y 0..11, then 12..23) keep LDS < 64 KB.
// Staged conv outputs kept in f32, channel-interleaved [pos][col*2+ntile]
// so each lane writes D as 8 packed ds_store_b64.
// avg_sq comes straight from the im2col slab (it IS the patch).
// ---------------------------------------------------------------------------
__global__ void conv_relu_pool_kernel(const float* __restrict__ in,     // [B,1,28,28]
                                      const float* __restrict__ cwmu,   // [32,25]
                                      const float* __restrict__ cwsig,  // [32]
                                      _Float16* __restrict__ mu4,       // [B,4608]
                                      _Float16* __restrict__ sg4,       // [B,4608]
                                      float* __restrict__ rsum) {       // [B]
  __shared__ float    simg[IMG];             // 3136 B
  __shared__ _Float16 sB[2 * 32 * 16];       // 2048 B  (2 N-tiles, B-operand layout)
  __shared__ _Float16 im2col[288 * 32];      // 18432 B (phase slab, K padded to 32)
  __shared__ float    mu1f[288 * 32];        // 36864 B (raw conv outs, [pos][col*2+nt])
  __shared__ float    avgsq[288];            // 1152 B
  __shared__ float    ssp[K_NUM];            // 128 B
  __shared__ float    sred[256];             // 1024 B
                                             // total ~62.8 KB

  const int b    = blockIdx.x;
  const int t    = threadIdx.x;
  const int lane = t & 31;
  const int wid  = __builtin_amdgcn_readfirstlane(t >> 5);  // scalar wave id

  const float* img = in + (size_t)b * IMG;
  for (int i = t; i < IMG; i += 256) simg[i] = img[i];
  // pack conv weights into B-operand layout: B[k][n] = w[n][k], zeros for k>=25
  for (int i = t; i < 2 * 32 * 16; i += 256) {
    int e  = i & 15;
    int ln = (i >> 4) & 31;
    int nt = i >> 9;
    int n  = nt * 16 + (ln & 15);
    int k  = b_klo(ln, e);
    sB[i] = (_Float16)((k < 25) ? cwmu[n * 25 + k] : 0.0f);
  }
  if (t < K_NUM) ssp[t] = softplusf(cwsig[t]);
  __syncthreads();

  const v16h bt0 = *(const v16h*)&sB[(0 * 32 + lane) * 16];
  const v16h bt1 = *(const v16h*)&sB[(1 * 32 + lane) * 16];
  const int  aoff = (lane < 16) ? 0 : 8;

  float racc = 0.0f;
  _Float16* mrow = mu4 + (size_t)b * FEAT;
  _Float16* srow = sg4 + (size_t)b * FEAT;

  for (int phase = 0; phase < 2; ++phase) {
    // ---- build f16 im2col slab for conv rows Y = phase*12 .. phase*12+11 ----
    for (int i = t; i < 288 * 32; i += 256) {
      int pos = i >> 5;            // 0..287 local position
      int k   = i & 31;
      int y   = pos / 24;
      int X   = pos - y * 24;
      int Y   = phase * 12 + y;
      float v = (k < 25) ? simg[(Y + k / 5) * 28 + (X + k % 5)] : 0.0f;
      im2col[i] = (_Float16)v;
    }
    __syncthreads();

    // ---- implicit-GEMM conv: 18 M-tiles split across 8 waves (scalar loop) ----
    for (int mt = wid; mt < 18; mt += 8) {
      int mrowi = mt * 16 + (lane & 15);
      v8h a0 = *(const v8h*)&im2col[mrowi * 32 + aoff];
      v8h a1 = *(const v8h*)&im2col[mrowi * 32 + 16 + aoff];
      v16h a = __builtin_shufflevector(a0, a1, 0,1,2,3,4,5,6,7,8,9,10,11,12,13,14,15);
      v8f c0 = {};
      v8f c1 = {};
      c0 = __builtin_amdgcn_wmma_f32_16x16x32_f16(false, a, false, bt0,
                                                  (short)0, c0, false, false);
      c1 = __builtin_amdgcn_wmma_f32_16x16x32_f16(false, a, false, bt1,
                                                  (short)0, c1, false, false);
      int col = lane & 15;
      #pragma unroll
      for (int v = 0; v < 8; ++v) {
        int pos = mt * 16 + ((lane < 16) ? v : v + 8);
        float2 p; p.x = c0[v]; p.y = c1[v];           // channels col (nt0), col+16 (nt1)
        *(float2*)&mu1f[pos * 32 + col * 2] = p;      // 8B-aligned ds_store_b64
      }
    }

    // ---- avg of squared patch entries per position (from im2col itself) ----
    for (int pos = t; pos < 288; pos += 256) {
      float s = 0.0f;
      #pragma unroll
      for (int k = 0; k < 25; ++k) {
        float v = (float)im2col[pos * 32 + k];
        s = fmaf(v, v, s);
      }
      avgsq[pos] = s * (1.0f / 25.0f);
    }
    __syncthreads();   // mu1f + avgsq ready

    // ---- ReLU + 2x2 maxpool + variance for pool rows phase*6 .. phase*6+5 ----
    for (int o = t; o < 2304; o += 256) {
      int c   = o / 72;
      int rem = o - c * 72;
      int pyl = rem / 12;
      int px  = rem - pyl * 12;
      int cidx = (c & 15) * 2 + (c >> 4);   // [pos][col*2+nt] channel remap

      float best = -1.0f;      // relu'd values >= 0: first window always beats init
      float bestRaw = 0.0f;
      int   bestPos = 0;
      #pragma unroll
      for (int k = 0; k < 4; ++k) {        // order (dy,dx) = reference flatten order
        int pos = (pyl * 2 + (k >> 1)) * 24 + px * 2 + (k & 1);
        float s = mu1f[pos * 32 + cidx];
        float rv = s > 0.0f ? s : 0.0f;
        if (rv > best) { best = rv; bestRaw = s; bestPos = pos; }   // strict: ties->first
      }
      float g = bestRaw > 0.0f ? 1.0f : 0.0f;
      float sigma = cleanf(g * ssp[c] * avgsq[bestPos]);

      int og = c * 144 + (phase * 6 + pyl) * 12 + px;
      mrow[og] = (_Float16)best;
      srow[og] = (_Float16)sigma;
      racc += best * best + sigma;
    }
    __syncthreads();   // before next phase overwrites im2col/mu1f
  }

  sred[t] = racc;
  __syncthreads();
  for (int s = 128; s > 0; s >>= 1) {
    if (t < s) sred[t] += sred[t + s];
    __syncthreads();
  }
  if (t == 0) rsum[b] = sred[0];
}

// ---------------------------------------------------------------------------
// Kernel 2: FC via WMMA.  mu5 = mu4 @ W, S1 = sigma4 @ W^2, N padded to 16.
// One wave per 16-row M tile, K = 4608 in 144 chunks of 32, f32 accumulate.
// ---------------------------------------------------------------------------
__global__ void fc_wmma_kernel(const _Float16* __restrict__ mu4,
                               const _Float16* __restrict__ sg4,
                               const _Float16* __restrict__ Bmu,
                               const _Float16* __restrict__ Bsq,
                               float* __restrict__ mu5,   // [B,16]
                               float* __restrict__ S1) {  // [B,16]
  const int wave = threadIdx.x >> 5;
  const int lane = threadIdx.x & 31;
  const int tile = blockIdx.x * (blockDim.x >> 5) + wave;

  const int row  = tile * 16 + (lane & 15);
  const _Float16* arow_mu = mu4 + (size_t)row * FEAT;
  const _Float16* arow_sg = sg4 + (size_t)row * FEAT;
  const int aoff = (lane < 16) ? 0 : 8;

  v8f cmu = {};
  v8f csg = {};

  for (int kc = 0; kc < KCHUNKS; ++kc) {
    const int kbase = kc * 32;
    if ((kc & 7) == 0 && kc + 8 < KCHUNKS) {   // gfx1250 global_prefetch_b8
      __builtin_prefetch(arow_mu + kbase + 256, 0, 1);
      __builtin_prefetch(arow_sg + kbase + 256, 0, 1);
    }
    v8h am0 = *(const v8h*)(arow_mu + kbase + aoff);
    v8h am1 = *(const v8h*)(arow_mu + kbase + 16 + aoff);
    v8h as0 = *(const v8h*)(arow_sg + kbase + aoff);
    v8h as1 = *(const v8h*)(arow_sg + kbase + 16 + aoff);
    v16h amu = __builtin_shufflevector(am0, am1, 0,1,2,3,4,5,6,7,8,9,10,11,12,13,14,15);
    v16h asg = __builtin_shufflevector(as0, as1, 0,1,2,3,4,5,6,7,8,9,10,11,12,13,14,15);
    v16h bmu = *(const v16h*)(Bmu + ((size_t)(kbase + lane)) * 16);
    v16h bsq = *(const v16h*)(Bsq + ((size_t)(kbase + lane)) * 16);

    cmu = __builtin_amdgcn_wmma_f32_16x16x32_f16(false, amu, false, bmu,
                                                 (short)0, cmu, false, false);
    csg = __builtin_amdgcn_wmma_f32_16x16x32_f16(false, asg, false, bsq,
                                                 (short)0, csg, false, false);
  }

  const int col = lane & 15;
  #pragma unroll
  for (int v = 0; v < 8; ++v) {
    int r = tile * 16 + ((lane < 16) ? v : v + 8);
    mu5[(size_t)r * 16 + col] = cmu[v];
    S1 [(size_t)r * 16 + col] = csg[v];
  }
}

// ---------------------------------------------------------------------------
// Kernel 3: softmax + variance propagation. One thread per image.
// ---------------------------------------------------------------------------
__global__ void finish_kernel(const float* __restrict__ mu5,
                              const float* __restrict__ S1,
                              const float* __restrict__ rsum,
                              const float* __restrict__ fcsig,
                              float* __restrict__ out, int B) {
  int b = blockIdx.x * blockDim.x + threadIdx.x;
  if (b >= B) return;
  float m[UNITS];
  float mx = -3.4e38f;
  #pragma unroll
  for (int u = 0; u < UNITS; ++u) { m[u] = mu5[(size_t)b * 16 + u]; mx = fmaxf(mx, m[u]); }
  float se = 0.0f;
  #pragma unroll
  for (int u = 0; u < UNITS; ++u) { m[u] = __expf(m[u] - mx); se += m[u]; }
  float inv = 1.0f / se;
  float r = rsum[b];
  #pragma unroll
  for (int u = 0; u < UNITS; ++u) {
    float p = m[u] * inv;
    out[(size_t)b * UNITS + u] = p;
    float Ws = softplusf(fcsig[u]);
    float s5 = cleanf(S1[(size_t)b * 16 + u] + r * Ws);
    float g = p - p * p;
    g = g * g;
    float so = cleanf(g * s5 * (1.0f / (float)UNITS));
    so = (so != so || __builtin_isinf(so)) ? 0.0f : so;
    out[(size_t)B * UNITS + (size_t)b * UNITS + u] = so;
  }
}

// ---------------------------------------------------------------------------
// Kernel 4: KL terms.  kl = 0.5*(sum w^2 + sum_units d*(s - 1 - log s))
// ---------------------------------------------------------------------------
__global__ void kl_kernel(const float* __restrict__ cwmu, const float* __restrict__ cwsig,
                          const float* __restrict__ fcw,  const float* __restrict__ fcsig,
                          float* __restrict__ out, int out_off) {
  __shared__ float red[256];
  int t = threadIdx.x;
  float acc = 0.0f;
  for (int i = t; i < K_NUM * 25; i += 256) acc += cwmu[i] * cwmu[i];
  for (int i = t; i < FEAT * UNITS; i += 256) acc += fcw[i] * fcw[i];
  if (t < K_NUM) { float s = softplusf(cwsig[t]); acc += 25.0f * (s - 1.0f - logf(s)); }
  if (t < UNITS) { float s = softplusf(fcsig[t]); acc += (float)FEAT * (s - 1.0f - logf(s)); }
  red[t] = acc;
  __syncthreads();
  for (int s = 128; s > 0; s >>= 1) {
    if (t < s) red[t] += red[t + s];
    __syncthreads();
  }
  if (t == 0) out[out_off] = 0.5f * red[0];
}

// ---------------------------------------------------------------------------
extern "C" void kernel_launch(void* const* d_in, const int* in_sizes, int n_in,
                              void* d_out, int out_size, void* d_ws, size_t ws_size,
                              hipStream_t stream) {
  const float* inputs     = (const float*)d_in[0];  // [B,1,28,28]
  const float* conv_w_mu  = (const float*)d_in[1];  // [32,1,5,5]
  const float* conv_w_sig = (const float*)d_in[2];  // [32]
  const float* fc_w_mu    = (const float*)d_in[3];  // [4608,10]
  const float* fc_w_sig   = (const float*)d_in[4];  // [10]
  float* out = (float*)d_out;

  const int B = in_sizes[0] / IMG;   // 2048
  const int tiles = B / 16;          // 128

  char* ws = (char*)d_ws;
  size_t off = 0;
  _Float16* Bmu = (_Float16*)(ws + off); off += (size_t)KCHUNKS * 32 * 16 * sizeof(_Float16);
  _Float16* Bsq = (_Float16*)(ws + off); off += (size_t)KCHUNKS * 32 * 16 * sizeof(_Float16);
  _Float16* mu4 = (_Float16*)(ws + off); off += (size_t)B * FEAT * sizeof(_Float16);
  _Float16* sg4 = (_Float16*)(ws + off); off += (size_t)B * FEAT * sizeof(_Float16);
  float* rsum   = (float*)(ws + off);    off += (size_t)B * sizeof(float);
  float* mu5    = (float*)(ws + off);    off += (size_t)B * 16 * sizeof(float);
  float* S1     = (float*)(ws + off);    off += (size_t)B * 16 * sizeof(float);

  pack_fc_kernel<<<64, 256, 0, stream>>>(fc_w_mu, Bmu, Bsq);
  conv_relu_pool_kernel<<<B, 256, 0, stream>>>(inputs, conv_w_mu, conv_w_sig,
                                               mu4, sg4, rsum);
  fc_wmma_kernel<<<tiles / 8, 256, 0, stream>>>(mu4, sg4, Bmu, Bsq, mu5, S1);
  finish_kernel<<<(B + 255) / 256, 256, 0, stream>>>(mu5, S1, rsum, fc_w_sig, out, B);
  kl_kernel<<<1, 256, 0, stream>>>(conv_w_mu, conv_w_sig, fc_w_mu, fc_w_sig,
                                   out, 2 * B * UNITS);
}
